// Accident_46084999086128
// MI455X (gfx1250) — matrix operand order repaired and checked by reading, not compiled, not run on hardware
//
#include <hip/hip_runtime.h>
#include <cstdint>

typedef __bf16 bf16_t;
typedef __attribute__((ext_vector_type(16))) __bf16 bf16x16;
typedef __attribute__((ext_vector_type(8)))  __bf16 bf16x8;
typedef __attribute__((ext_vector_type(8)))  float  floatx8;

// A fragment: elems k..k+7 and k+16..k+23 (ISA 16-bit A 16x32 layout, per lane half)
static __device__ __forceinline__ bf16x16 load_afrag(const bf16_t* p) {
  bf16x8 lo = *(const bf16x8*)p;
  bf16x8 hi = *(const bf16x8*)(p + 16);
  return __builtin_shufflevector(lo, hi, 0,1,2,3,4,5,6,7,8,9,10,11,12,13,14,15);
}
// B fragment: elems k..k+7 and k+16..k+23 (same layout as A, per lane half)
static __device__ __forceinline__ bf16x16 load_bfrag(const bf16_t* p) {
  bf16x8 lo = *(const bf16x8*)p;
  bf16x8 hi = *(const bf16x8*)(p + 16);
  return __builtin_shufflevector(lo, hi, 0,1,2,3,4,5,6,7,8,9,10,11,12,13,14,15);
}

// ============================================================
// Batched GEMM: C = act(alpha * A@B + bias) + resid
// A bf16 [M,K] row major (lda). B in K-contiguous ("transposed") layout:
// B[k,n] = Bptr[n*ldb + k]. batch bi = b*nH + h with per-matrix strides.
// Wave computes a 32x64 tile: 8x v_wmma_f32_16x16x32_bf16 per K-step.
// All fragment loads are 16B-aligned vector loads.
// ============================================================
struct GemmParams {
  const bf16_t* A;
  const bf16_t* B;
  float*  Cf;
  bf16_t* Cb;
  const float* bias;
  const float* resid;
  int M, N, K;
  int lda, ldb, ldc;
  long long asb, ash, bsb, bsh, csb, csh;
  int nH, act;              // act: 0=none, 1=gelu(tanh)
  float alpha;
  int tilesM, tilesN;
};

__global__ __launch_bounds__(128) void gemm_bf16_kernel(GemmParams p) {
  const int lane = threadIdx.x & 31;
  const int wid  = blockIdx.x * 4 + (threadIdx.x >> 5);
  if (wid >= p.tilesM * p.tilesN) return;
  const int tm = wid / p.tilesN;
  const int tn = wid % p.tilesN;
  const int bi = blockIdx.y;
  const int b  = bi / p.nH;
  const int h  = bi % p.nH;

  const bf16_t* Ab = p.A + (long long)b * p.asb + (long long)h * p.ash;
  const bf16_t* Bb = p.B + (long long)b * p.bsb + (long long)h * p.bsh;

  const int m0 = tm * 32, n0 = tn * 64;
  const int half = lane >> 4;                 // 0 or 1

  int row0 = m0 + (lane & 15);       if (row0 > p.M - 1) row0 = p.M - 1;
  int row1 = m0 + 16 + (lane & 15);  if (row1 > p.M - 1) row1 = p.M - 1;
  const bf16_t* arow0 = Ab + (long long)row0 * p.lda + half * 8;
  const bf16_t* arow1 = Ab + (long long)row1 * p.lda + half * 8;

  // per-tile B column pointers (hoisted out of the K loop)
  const bf16_t* bptr[4];
#pragma unroll
  for (int t = 0; t < 4; ++t) {
    int n = n0 + t * 16 + (lane & 15); if (n > p.N - 1) n = p.N - 1;
    bptr[t] = Bb + (long long)n * p.ldb;
  }

  floatx8 z = {0.f,0.f,0.f,0.f,0.f,0.f,0.f,0.f};
  floatx8 acc[2][4];
#pragma unroll
  for (int mi = 0; mi < 2; ++mi)
#pragma unroll
    for (int t = 0; t < 4; ++t) acc[mi][t] = z;

  for (int k0 = 0; k0 < p.K; k0 += 32) {
    const bf16x16 af0 = load_afrag(arow0 + k0);
    const bf16x16 af1 = load_afrag(arow1 + k0);
    const int kb = k0 + half * 8;
#pragma unroll
    for (int t = 0; t < 4; ++t) {
      const bf16x16 bfv = load_bfrag(bptr[t] + kb);
      acc[0][t] = __builtin_amdgcn_wmma_f32_16x16x32_bf16(false, af0, false, bfv,
                                                          (short)0, acc[0][t], false, false);
      acc[1][t] = __builtin_amdgcn_wmma_f32_16x16x32_bf16(false, af1, false, bfv,
                                                          (short)0, acc[1][t], false, false);
    }
  }

#pragma unroll
  for (int mi = 0; mi < 2; ++mi) {
    const int mbase = m0 + mi * 16 + half * 8;
#pragma unroll
    for (int t = 0; t < 4; ++t) {
      const int n = n0 + t * 16 + (lane & 15);
      if (n >= p.N) continue;
#pragma unroll
      for (int j = 0; j < 8; ++j) {
        const int m = mbase + j;
        if (m >= p.M) continue;
        float v = acc[mi][t][j] * p.alpha;
        if (p.bias) v += p.bias[n];
        if (p.act == 1) {
          float u = v;
          v = 0.5f * u * (1.f + tanhf(0.7978845608028654f * (u + 0.044715f * u * u * u)));
        }
        const long long coff = (long long)b * p.csb + (long long)h * p.csh +
                               (long long)m * p.ldc + n;
        if (p.resid) v += p.resid[coff];
        if (p.Cf) p.Cf[coff] = v;
        if (p.Cb) p.Cb[coff] = (bf16_t)v;
      }
    }
  }
}

// ================= LayerNorm over D=192, wave per row =================
__global__ __launch_bounds__(128) void ln192_kernel(const float* __restrict__ x,
    const float* __restrict__ g, const float* __restrict__ b,
    float* of, bf16_t* ob, int rows) {
  const int lane = threadIdx.x & 31;
  const int row  = blockIdx.x * 4 + (threadIdx.x >> 5);
  if (row >= rows) return;
  const float* xr = x + (long long)row * 192;
  float v[6], s = 0.f, s2 = 0.f;
#pragma unroll
  for (int i = 0; i < 6; ++i) { float t = xr[lane + 32 * i]; v[i] = t; s += t; s2 += t * t; }
#pragma unroll
  for (int o = 16; o >= 1; o >>= 1) { s += __shfl_xor(s, o, 32); s2 += __shfl_xor(s2, o, 32); }
  const float mean = s * (1.f / 192.f);
  const float var  = s2 * (1.f / 192.f) - mean * mean;
  const float inv  = rsqrtf(var + 1e-6f);
#pragma unroll
  for (int i = 0; i < 6; ++i) {
    const int d = lane + 32 * i;
    const float o = (v[i] - mean) * inv * g[d] + b[d];
    if (of) of[(long long)row * 192 + d] = o;
    if (ob) ob[(long long)row * 192 + d] = (bf16_t)o;
  }
}

// ====== Softmax: f32 rows [rows, Lpad]; writes bf16 probs aliased in-place
//        (bf16 row stride = 2*Lpad elems = same byte offset as the f32 row) ======
__global__ __launch_bounds__(128) void softmax_kernel(float* sc, int rows, int L, int Lpad) {
  const int lane = threadIdx.x & 31;
  const int row  = blockIdx.x * 4 + (threadIdx.x >> 5);
  if (row >= rows) return;
  const float* xr = sc + (long long)row * Lpad;
  bf16_t* pr = (bf16_t*)sc + (long long)row * (2 * Lpad);
  const int EPB = Lpad / 32;   // <= 7
  float v[7], mx = -3.4e38f;
  for (int i = 0; i < EPB; ++i) {
    const int e = lane + 32 * i;
    const float t = (e < L) ? xr[e] : -3.4e38f;
    v[i] = t; mx = fmaxf(mx, t);
  }
  for (int o = 16; o >= 1; o >>= 1) mx = fmaxf(mx, __shfl_xor(mx, o, 32));
  float sum = 0.f;
  for (int i = 0; i < EPB; ++i) {
    const int e = lane + 32 * i;
    v[i] = (e < L) ? __expf(v[i] - mx) : 0.f;
    sum += v[i];
  }
  for (int o = 16; o >= 1; o >>= 1) sum += __shfl_xor(sum, o, 32);
  const float r = 1.f / sum;
  for (int i = 0; i < EPB; ++i) {
    const int e = lane + 32 * i;
    if (e < Lpad) pr[e] = (bf16_t)(v[i] * r);
  }
}

// ====== Patch unfold: x[BT,3,224,224] -> patches bf16 [BT,196,768] ======
__global__ void unfold_kernel(const float* __restrict__ x, bf16_t* out, long long total) {
  long long i = (long long)blockIdx.x * blockDim.x + threadIdx.x;
  if (i >= total) return;
  const int c = (int)(i % 768); long long r = i / 768;
  const int p = (int)(r % 196); const int bt = (int)(r / 196);
  const int ch = c >> 8, py = (c >> 4) & 15, px = c & 15;
  const int ph = p / 14, pw = p % 14;
  out[i] = (bf16_t)x[(((long long)bt * 3 + ch) * 224 + ph * 16 + py) * 224 + pw * 16 + px];
}

// ====== cls token + pos-emb: X[B*T,197,192] ======
__global__ void addclspos_kernel(float* X, const float* __restrict__ pos,
                                 const float* __restrict__ cls, int T, long long total) {
  long long i = (long long)blockIdx.x * blockDim.x + threadIdx.x;
  if (i >= total) return;
  const int d = (int)(i % 192); long long r = i / 192;
  const int n = (int)(r % 197); const int bt = (int)(r / 197);
  const int t = bt % T;
  const float base = (n == 0) ? cls[d] : X[i];
  X[i] = base + pos[((long long)t * 197 + n) * 192 + d];
}

// ====== transpose-convert weights: W f32 [L][K][N] -> WT bf16 [L][N][K] ======
__global__ void cvt_t_kernel(const float* __restrict__ w, bf16_t* wt,
                             int K, int N, long long total) {
  long long i = (long long)blockIdx.x * blockDim.x + threadIdx.x;
  if (i >= total) return;
  const int k = (int)(i % K); long long r = i / K;
  const int n = (int)(r % N); const long long l = r / N;
  wt[i] = (bf16_t)w[(l * K + k) * N + n];
}

// ====== transpose V out of fused qkv: VT[b][h][d][token], zero-padded tokens ======
__global__ void transpose_v_kernel(const bf16_t* __restrict__ qkv, int ldq, int voff, int dh,
                                   int N, int Npad, int H, bf16_t* vt, long long total) {
  long long i = (long long)blockIdx.x * blockDim.x + threadIdx.x;
  if (i >= total) return;
  const int k = (int)(i % Npad); long long r = i / Npad;
  const int d = (int)(r % dh); r /= dh;
  const int h = (int)(r % H); const int b = (int)(r / H);
  bf16_t v = (bf16_t)0.f;
  if (k < N) v = qkv[(long long)b * N * ldq + (long long)k * ldq + voff + h * dh + d];
  vt[i] = v;
}

// ====== ATS significance + inverse-transform sampling, one block per batch ======
__global__ __launch_bounds__(256) void ats_kernel(const bf16_t* __restrict__ probs, int prow,
    const bf16_t* __restrict__ qkv, int ldq, int voff, int dh, int H, int N, int* idxout) {
  const int b = blockIdx.x, tid = threadIdx.x, M = N - 1;
  __shared__ float s[256];
  __shared__ float cdf[256];
  float sv = 0.f;
  if (tid < M) {
    for (int h = 0; h < H; ++h) {
      const float pcls = (float)probs[((long long)(b * H + h) * N + 0) * prow + (1 + tid)];
      const bf16_t* vp = qkv + (long long)b * N * ldq + (long long)(1 + tid) * ldq + voff + h * dh;
      float nn = 0.f;
      for (int d = 0; d < dh; ++d) { const float xv = (float)vp[d]; nn += xv * xv; }
      sv += pcls * sqrtf(nn);
    }
    sv /= (float)H;
  }
  s[tid] = (tid < M) ? sv : 0.f;
  __syncthreads();
  if (tid == 0) {
    float tot = 0.f;
    for (int j = 0; j < M; ++j) tot += s[j];
    tot += 1e-8f;
    float c = 0.f;
    for (int j = 0; j < M; ++j) { c += s[j] / tot; cdf[j] = c; }
  }
  __syncthreads();
  if (tid < M) {
    const float u = ((float)tid + 0.5f) / (float)M;
    int lo = 0, hi = M;
    while (lo < hi) { const int mid = (lo + hi) >> 1; if (cdf[mid] < u) lo = mid + 1; else hi = mid; }
    int id = lo; if (id > M - 1) id = M - 1;   // clip to [0, N-2]
    idxout[b * N + 1 + tid] = id + 1;
  }
  if (tid == 0) idxout[b * N] = 0;
}

// ====== gather-add: Xn[bt,j,:] = X[bt,idx,:] + Y[bt,idx,:] ======
__global__ void gather_add_kernel(const float* __restrict__ X, const float* __restrict__ Y,
                                  const int* __restrict__ idx, float* Xn, long long total) {
  long long i = (long long)blockIdx.x * blockDim.x + threadIdx.x;
  if (i >= total) return;
  const int d = (int)(i % 192); long long r = i / 192;
  const int j = (int)(r % 197); const int bt = (int)(r / 197);
  const int id = idx[bt * 197 + j];
  const long long src = ((long long)bt * 197 + id) * 192 + d;
  Xn[i] = X[src] + Y[src];
}

// ====== extract cls per frame + temporal token: XC [B, T+1, 192] ======
__global__ void extract_cls_kernel(const float* __restrict__ X, const float* __restrict__ ttok,
                                   float* XC, int T, long long total) {
  long long i = (long long)blockIdx.x * blockDim.x + threadIdx.x;
  if (i >= total) return;
  const int d = (int)(i % 192); long long r = i / 192;
  const int n = (int)(r % (T + 1)); const int b = (int)(r / (T + 1));
  XC[i] = (n == 0) ? ttok[d]
                   : X[((long long)(b * T + (n - 1)) * 197 + 0) * 192 + d];
}

// ====== head: maxpool over tokens -> LN(final) -> relu(f@h1+b)@h2+b ======
__global__ __launch_bounds__(192) void head_kernel(const float* __restrict__ XC,
    const float* __restrict__ fg, const float* __restrict__ fb,
    const float* __restrict__ h1w, const float* __restrict__ h1b,
    const float* __restrict__ h2w, const float* __restrict__ h2b,
    float* out, int Ntok) {
  const int b = blockIdx.x, tid = threadIdx.x;
  __shared__ float f[192];
  __shared__ float hid[64];
  __shared__ float stats[2];
  float m = -3.4e38f;
  for (int n = 0; n < Ntok; ++n)
    m = fmaxf(m, XC[((long long)b * Ntok + n) * 192 + tid]);
  f[tid] = m;
  __syncthreads();
  if (tid == 0) {
    float s = 0.f, s2 = 0.f;
    for (int d = 0; d < 192; ++d) { s += f[d]; s2 += f[d] * f[d]; }
    const float mean = s / 192.f;
    stats[0] = mean;
    stats[1] = rsqrtf(s2 / 192.f - mean * mean + 1e-6f);
  }
  __syncthreads();
  const float fn = (m - stats[0]) * stats[1] * fg[tid] + fb[tid];
  __syncthreads();
  f[tid] = fn;
  __syncthreads();
  if (tid < 64) {
    float a = h1b[tid];
    for (int d = 0; d < 192; ++d) a += f[d] * h1w[d * 64 + tid];
    hid[tid] = fmaxf(a, 0.f);
  }
  __syncthreads();
  if (tid < 2) {
    float a = h2b[tid];
    for (int d = 0; d < 64; ++d) a += hid[d] * h2w[d * 2 + tid];
    out[b * 2 + tid] = a;
  }
}

// ============================================================
// Host orchestration
// ============================================================
static void launch_gemm(hipStream_t st,
    const bf16_t* A, const bf16_t* Bm, float* Cf, bf16_t* Cb,
    const float* bias, const float* resid,
    int M, int N, int K,
    int lda, int ldb, int ldc,
    long long asb, long long ash, long long bsb, long long bsh,
    long long csb, long long csh,
    int batches, int nH, int act, float alpha) {
  GemmParams p;
  p.A = A; p.B = Bm; p.Cf = Cf; p.Cb = Cb; p.bias = bias; p.resid = resid;
  p.M = M; p.N = N; p.K = K; p.lda = lda; p.ldb = ldb; p.ldc = ldc;
  p.asb = asb; p.ash = ash; p.bsb = bsb; p.bsh = bsh; p.csb = csb; p.csh = csh;
  p.nH = nH; p.act = act; p.alpha = alpha;
  p.tilesM = (M + 31) / 32; p.tilesN = (N + 63) / 64;
  const int waves = p.tilesM * p.tilesN;
  dim3 grid((unsigned)((waves + 3) / 4), (unsigned)batches);
  gemm_bf16_kernel<<<grid, 128, 0, st>>>(p);
}

extern "C" void kernel_launch(void* const* d_in, const int* in_sizes, int n_in,
                              void* d_out, int out_size, void* d_ws, size_t ws_size,
                              hipStream_t stream) {
  (void)in_sizes; (void)n_in; (void)out_size; (void)ws_size;
  const float* xin      = (const float*)d_in[0];
  const float* patch_w  = (const float*)d_in[1];
  const float* patch_b  = (const float*)d_in[2];
  const float* pos_emb  = (const float*)d_in[3];
  const float* space_tk = (const float*)d_in[4];
  const float* temp_tk  = (const float*)d_in[5];
  const float* s_ln1_g = (const float*)d_in[6];  const float* s_ln1_b = (const float*)d_in[7];
  const float* s_qkv_w = (const float*)d_in[8];  const float* s_out_w = (const float*)d_in[9];
  const float* s_out_b = (const float*)d_in[10];
  const float* s_ln2_g = (const float*)d_in[11]; const float* s_ln2_b = (const float*)d_in[12];
  const float* s_fc1_w = (const float*)d_in[13]; const float* s_fc1_b = (const float*)d_in[14];
  const float* s_fc2_w = (const float*)d_in[15]; const float* s_fc2_b = (const float*)d_in[16];
  const float* s_nrm_g = (const float*)d_in[17]; const float* s_nrm_b = (const float*)d_in[18];
  const float* t_ln1_g = (const float*)d_in[19]; const float* t_ln1_b = (const float*)d_in[20];
  const float* t_qkv_w = (const float*)d_in[21]; const float* t_out_w = (const float*)d_in[22];
  const float* t_out_b = (const float*)d_in[23];
  const float* t_ln2_g = (const float*)d_in[24]; const float* t_ln2_b = (const float*)d_in[25];
  const float* t_fc1_w = (const float*)d_in[26]; const float* t_fc1_b = (const float*)d_in[27];
  const float* t_fc2_w = (const float*)d_in[28]; const float* t_fc2_b = (const float*)d_in[29];
  const float* t_nrm_g = (const float*)d_in[30]; const float* t_nrm_b = (const float*)d_in[31];
  const float* a_ln1_g = (const float*)d_in[32]; const float* a_ln1_b = (const float*)d_in[33];
  const float* a_qkv_w = (const float*)d_in[34]; const float* a_qkv_b = (const float*)d_in[35];
  const float* a_prj_w = (const float*)d_in[36]; const float* a_prj_b = (const float*)d_in[37];
  const float* a_ln2_g = (const float*)d_in[38]; const float* a_ln2_b = (const float*)d_in[39];
  const float* a_fc1_w = (const float*)d_in[40]; const float* a_fc1_b = (const float*)d_in[41];
  const float* a_fc2_w = (const float*)d_in[42]; const float* a_fc2_b = (const float*)d_in[43];
  const float* final_g = (const float*)d_in[44]; const float* final_b = (const float*)d_in[45];
  const float* h1_w = (const float*)d_in[46];    const float* h1_b = (const float*)d_in[47];
  const float* h2_w = (const float*)d_in[48];    const float* h2_b = (const float*)d_in[49];
  float* out = (float*)d_out;

  const int B = 16, T = 5, BT = 80, N = 197, H = 6;
  const int ROWS = BT * N;                     // 15760
  const int NPAD = 224;                        // 197 -> next mult of 32
  hipStream_t st = stream;

  // ---- workspace bump allocator ----
  char* base = (char*)d_ws; size_t off = 0;
  auto alloc = [&](size_t bytes) -> void* {
    void* p = base + off; off += bytes; off = (off + 255) & ~(size_t)255; return p;
  };
  // bf16 transposed weights  WT[l][n][k]
  bf16_t* Wpatch = (bf16_t*)alloc(sizeof(bf16_t) * 768 * 192);
  bf16_t* Wsqkv  = (bf16_t*)alloc(sizeof(bf16_t) * 2 * 192 * 1152);
  bf16_t* Wsout  = (bf16_t*)alloc(sizeof(bf16_t) * 2 * 384 * 192);
  bf16_t* Wsfc1  = (bf16_t*)alloc(sizeof(bf16_t) * 2 * 192 * 768);
  bf16_t* Wsfc2  = (bf16_t*)alloc(sizeof(bf16_t) * 2 * 768 * 192);
  bf16_t* Wtqkv  = (bf16_t*)alloc(sizeof(bf16_t) * 3 * 192 * 1152);
  bf16_t* Wtout  = (bf16_t*)alloc(sizeof(bf16_t) * 3 * 384 * 192);
  bf16_t* Wtfc1  = (bf16_t*)alloc(sizeof(bf16_t) * 3 * 192 * 768);
  bf16_t* Wtfc2  = (bf16_t*)alloc(sizeof(bf16_t) * 3 * 768 * 192);
  bf16_t* Waqkv  = (bf16_t*)alloc(sizeof(bf16_t) * 12 * 192 * 576);
  bf16_t* Waprj  = (bf16_t*)alloc(sizeof(bf16_t) * 12 * 192 * 192);
  bf16_t* Wafc1  = (bf16_t*)alloc(sizeof(bf16_t) * 12 * 192 * 768);
  bf16_t* Wafc2  = (bf16_t*)alloc(sizeof(bf16_t) * 12 * 768 * 192);
  // activations
  bf16_t* PATCH = (bf16_t*)alloc(sizeof(bf16_t) * (size_t)BT * 196 * 768);
  float*  X     = (float*) alloc(sizeof(float)  * (size_t)ROWS * 192);
  float*  XN    = (float*) alloc(sizeof(float)  * (size_t)ROWS * 192);
  float*  Y     = (float*) alloc(sizeof(float)  * (size_t)ROWS * 192);
  bf16_t* LNB   = (bf16_t*)alloc(sizeof(bf16_t) * (size_t)ROWS * 192);
  bf16_t* QKV   = (bf16_t*)alloc(sizeof(bf16_t) * (size_t)ROWS * 1152);
  float*  SC    = (float*) alloc(sizeof(float)  * (size_t)BT * H * N * NPAD);
  bf16_t* VT    = (bf16_t*)alloc(sizeof(bf16_t) * (size_t)BT * H * 64 * NPAD);
  bf16_t* O     = (bf16_t*)alloc(sizeof(bf16_t) * (size_t)ROWS * 384);
  bf16_t* H1    = (bf16_t*)alloc(sizeof(bf16_t) * (size_t)ROWS * 768);
  int*    IDX   = (int*)   alloc(sizeof(int)    * (size_t)ROWS);
  float*  XC    = (float*) alloc(sizeof(float)  * (size_t)B * 6 * 192);
  bf16_t* LNC   = (bf16_t*)alloc(sizeof(bf16_t) * (size_t)B * 6 * 192);
  bf16_t* QKVC  = (bf16_t*)alloc(sizeof(bf16_t) * (size_t)B * 6 * 1152);
  float*  SCC   = (float*) alloc(sizeof(float)  * (size_t)B * H * 6 * 32);
  bf16_t* OC    = (bf16_t*)alloc(sizeof(bf16_t) * (size_t)B * 6 * 384);
  bf16_t* H1C   = (bf16_t*)alloc(sizeof(bf16_t) * (size_t)B * 6 * 768);

  // transpose-convert all weights: [L][K][N] f32 -> [L][N][K] bf16
  auto cvtt = [&](const float* src, bf16_t* dst, int K, int Nw, long long L) {
    long long tot = L * K * Nw;
    cvt_t_kernel<<<dim3((unsigned)((tot + 255) / 256)), 256, 0, st>>>(src, dst, K, Nw, tot);
  };
  cvtt(patch_w, Wpatch, 768, 192, 1);
  cvtt(s_qkv_w, Wsqkv, 192, 1152, 2); cvtt(s_out_w, Wsout, 384, 192, 2);
  cvtt(s_fc1_w, Wsfc1, 192, 768, 2);  cvtt(s_fc2_w, Wsfc2, 768, 192, 2);
  cvtt(t_qkv_w, Wtqkv, 192, 1152, 3); cvtt(t_out_w, Wtout, 384, 192, 3);
  cvtt(t_fc1_w, Wtfc1, 192, 768, 3);  cvtt(t_fc2_w, Wtfc2, 768, 192, 3);
  cvtt(a_qkv_w, Waqkv, 192, 576, 12); cvtt(a_prj_w, Waprj, 192, 192, 12);
  cvtt(a_fc1_w, Wafc1, 192, 768, 12); cvtt(a_fc2_w, Wafc2, 768, 192, 12);

  auto ln = [&](const float* xp, const float* g, const float* b, float* of, bf16_t* ob, int rows) {
    ln192_kernel<<<dim3((unsigned)((rows + 3) / 4)), 128, 0, st>>>(xp, g, b, of, ob, rows);
  };
  auto sm = [&](float* sc, int rows, int L, int Lpad) {
    softmax_kernel<<<dim3((unsigned)((rows + 3) / 4)), 128, 0, st>>>(sc, rows, L, Lpad);
  };
  auto trv = [&](const bf16_t* qkv, int ldq, int voff, int dh, int Ntok, int Npad, bf16_t* vt) {
    long long tot = (long long)(ldq == 1152 && Ntok == 6 ? B : BT) * H * dh * Npad;
    // batches dim = qkv rows / Ntok; spatial/ATS: BT, temporal: B
    transpose_v_kernel<<<dim3((unsigned)((tot + 255) / 256)), 256, 0, st>>>(
        qkv, ldq, voff, dh, Ntok, Npad, H, vt, tot);
  };

  // ---- patch embedding ----
  {
    long long tot = (long long)BT * 196 * 768;
    unfold_kernel<<<dim3((unsigned)((tot + 255) / 256)), 256, 0, st>>>(xin, PATCH, tot);
    launch_gemm(st, PATCH, Wpatch, X + 192, nullptr, patch_b, nullptr,
                196, 192, 768, 768, 768, 192,
                196LL * 768, 0, 0, 0, 197LL * 192, 0, BT, 1, 0, 1.f);
    long long t2 = (long long)ROWS * 192;
    addclspos_kernel<<<dim3((unsigned)((t2 + 255) / 256)), 256, 0, st>>>(X, pos_emb, space_tk, T, t2);
  }

  // ---- spatial transformer (2 layers, dh=64, inner 384) ----
  for (int i = 0; i < 2; ++i) {
    ln(X, s_ln1_g + i * 192, s_ln1_b + i * 192, nullptr, LNB, ROWS);
    launch_gemm(st, LNB, Wsqkv + (long long)i * 192 * 1152, nullptr, QKV, nullptr, nullptr,
                ROWS, 1152, 192, 192, 192, 1152, 0, 0, 0, 0, 0, 0, 1, 1, 0, 1.f);
    launch_gemm(st, QKV, QKV + 384, SC, nullptr, nullptr, nullptr,          // Q @ K^T
                N, N, 64, 1152, 1152, NPAD,
                (long long)N * 1152, 64, (long long)N * 1152, 64,
                (long long)H * N * NPAD, (long long)N * NPAD, BT * H, H, 0, 0.125f);
    sm(SC, BT * H * N, N, NPAD);
    trv(QKV, 1152, 768, 64, N, NPAD, VT);
    launch_gemm(st, (bf16_t*)SC, VT, nullptr, O, nullptr, nullptr,          // P @ V
                N, 64, NPAD, 2 * NPAD, NPAD, 384,
                (long long)H * N * 2 * NPAD, (long long)N * 2 * NPAD,
                (long long)H * 64 * NPAD, 64LL * NPAD,
                (long long)N * 384, 64, BT * H, H, 0, 1.f);
    launch_gemm(st, O, Wsout + (long long)i * 192 * 384, X, nullptr,
                s_out_b + i * 192, X,
                ROWS, 192, 384, 384, 384, 192, 0, 0, 0, 0, 0, 0, 1, 1, 0, 1.f);
    ln(X, s_ln2_g + i * 192, s_ln2_b + i * 192, nullptr, LNB, ROWS);
    launch_gemm(st, LNB, Wsfc1 + (long long)i * 192 * 768, nullptr, H1,
                s_fc1_b + i * 768, nullptr,
                ROWS, 768, 192, 192, 192, 768, 0, 0, 0, 0, 0, 0, 1, 1, 1, 1.f);
    launch_gemm(st, H1, Wsfc2 + (long long)i * 768 * 192, X, nullptr,
                s_fc2_b + i * 192, X,
                ROWS, 192, 768, 768, 768, 192, 0, 0, 0, 0, 0, 0, 1, 1, 0, 1.f);
  }
  ln(X, s_nrm_g, s_nrm_b, X, nullptr, ROWS);

  // ---- 12 ATS blocks (dh=32) ----
  float* xcur = X; float* xalt = XN;
  for (int i = 0; i < 12; ++i) {
    ln(xcur, a_ln1_g + i * 192, a_ln1_b + i * 192, nullptr, LNB, ROWS);
    launch_gemm(st, LNB, Waqkv + (long long)i * 192 * 576, nullptr, QKV,
                a_qkv_b + i * 576, nullptr,
                ROWS, 576, 192, 192, 192, 576, 0, 0, 0, 0, 0, 0, 1, 1, 0, 1.f);
    launch_gemm(st, QKV, QKV + 192, SC, nullptr, nullptr, nullptr,
                N, N, 32, 576, 576, NPAD,
                (long long)N * 576, 32, (long long)N * 576, 32,
                (long long)H * N * NPAD, (long long)N * NPAD, BT * H, H, 0, 0.17677669529663688f);
    sm(SC, BT * H * N, N, NPAD);
    trv(QKV, 576, 384, 32, N, NPAD, VT);
    launch_gemm(st, (bf16_t*)SC, VT, nullptr, O, nullptr, nullptr,
                N, 32, NPAD, 2 * NPAD, NPAD, 192,
                (long long)H * N * 2 * NPAD, (long long)N * 2 * NPAD,
                (long long)H * 32 * NPAD, 32LL * NPAD,
                (long long)N * 192, 32, BT * H, H, 0, 1.f);
    launch_gemm(st, O, Waprj + (long long)i * 192 * 192, Y, nullptr,
                a_prj_b + i * 192, nullptr,
                ROWS, 192, 192, 192, 192, 192, 0, 0, 0, 0, 0, 0, 1, 1, 0, 1.f);
    ats_kernel<<<dim3((unsigned)BT), 256, 0, st>>>((bf16_t*)SC, 2 * NPAD, QKV, 576, 384, 32, H, N, IDX);
    {
      long long tot = (long long)ROWS * 192;
      gather_add_kernel<<<dim3((unsigned)((tot + 255) / 256)), 256, 0, st>>>(xcur, Y, IDX, xalt, tot);
      float* tmp = xcur; xcur = xalt; xalt = tmp;
    }
    ln(xcur, a_ln2_g + i * 192, a_ln2_b + i * 192, nullptr, LNB, ROWS);
    launch_gemm(st, LNB, Wafc1 + (long long)i * 192 * 768, nullptr, H1,
                a_fc1_b + i * 768, nullptr,
                ROWS, 768, 192, 192, 192, 768, 0, 0, 0, 0, 0, 0, 1, 1, 1, 1.f);
    launch_gemm(st, H1, Wafc2 + (long long)i * 768 * 192, xcur, nullptr,
                a_fc2_b + i * 192, xcur,
                ROWS, 192, 768, 768, 768, 192, 0, 0, 0, 0, 0, 0, 1, 1, 0, 1.f);
  }

  // ---- temporal transformer over [B, 6, 192] ----
  {
    long long tot = (long long)B * 6 * 192;
    extract_cls_kernel<<<dim3((unsigned)((tot + 255) / 256)), 256, 0, st>>>(xcur, temp_tk, XC, T, tot);
  }
  const int TR = B * 6;      // 96 rows
  for (int i = 0; i < 3; ++i) {
    ln(XC, t_ln1_g + i * 192, t_ln1_b + i * 192, nullptr, LNC, TR);
    launch_gemm(st, LNC, Wtqkv + (long long)i * 192 * 1152, nullptr, QKVC, nullptr, nullptr,
                TR, 1152, 192, 192, 192, 1152, 0, 0, 0, 0, 0, 0, 1, 1, 0, 1.f);
    launch_gemm(st, QKVC, QKVC + 384, SCC, nullptr, nullptr, nullptr,
                6, 6, 64, 1152, 1152, 32,
                6LL * 1152, 64, 6LL * 1152, 64,
                (long long)H * 6 * 32, 6LL * 32, B * H, H, 0, 0.125f);
    sm(SCC, B * H * 6, 6, 32);
    trv(QKVC, 1152, 768, 64, 6, 32, VT);
    launch_gemm(st, (bf16_t*)SCC, VT, nullptr, OC, nullptr, nullptr,
                6, 64, 32, 64, 32, 384,
                (long long)H * 6 * 64, 6LL * 64,
                (long long)H * 64 * 32, 64LL * 32,
                6LL * 384, 64, B * H, H, 0, 1.f);
    launch_gemm(st, OC, Wtout + (long long)i * 192 * 384, XC, nullptr,
                t_out_b + i * 192, XC,
                TR, 192, 384, 384, 384, 192, 0, 0, 0, 0, 0, 0, 1, 1, 0, 1.f);
    ln(XC, t_ln2_g + i * 192, t_ln2_b + i * 192, nullptr, LNC, TR);
    launch_gemm(st, LNC, Wtfc1 + (long long)i * 192 * 768, nullptr, H1C,
                t_fc1_b + i * 768, nullptr,
                TR, 768, 192, 192, 192, 768, 0, 0, 0, 0, 0, 0, 1, 1, 1, 1.f);
    launch_gemm(st, H1C, Wtfc2 + (long long)i * 768 * 192, XC, nullptr,
                t_fc2_b + i * 192, XC,
                TR, 192, 768, 768, 768, 192, 0, 0, 0, 0, 0, 0, 1, 1, 0, 1.f);
  }
  ln(XC, t_nrm_g, t_nrm_b, XC, nullptr, TR);

  // ---- maxpool + final LN + MLP head ----
  head_kernel<<<dim3((unsigned)B), 192, 0, st>>>(XC, final_g, final_b, h1_w, h1_b, h2_w, h2_b, out, 6);
}
